// lstmAE_feedback_68624987456252
// MI455X (gfx1250) — matrix operand order repaired and checked by reading, not compiled, not run on hardware
//
#include <hip/hip_runtime.h>

#define BB 512
#define TT 256
#define DIM 128
#define DZ 256
#define HH 512
#define G4 2048

typedef __attribute__((ext_vector_type(16))) _Float16 v16h;
typedef __attribute__((ext_vector_type(8)))  _Float16 v8h;
typedef __attribute__((ext_vector_type(8)))  float    v8f;

static __device__ __forceinline__ v8f wmma16(v16h a, v16h b, v8f c) {
  // 8 args: (neg_a, A, neg_b, B, c_mod, C, reuse_a, reuse_b)
  return __builtin_amdgcn_wmma_f32_16x16x32_f16(false, a, false, b, (short)0, c, false, false);
}

// A fragment (16x32 f16): lanes 0-15 row m=lane, K range [k0+sel*16, +16) contiguous
static __device__ __forceinline__ v16h load_a_frag(const _Float16* base, int strideH, int k0, int lane) {
  int m   = lane & 15;
  int sel = lane >> 4;
  const _Float16* p = base + (size_t)m * strideH + k0 + sel * 16;
  union { v16h v; v8h h[2]; } u;
  u.h[0] = *(const v8h*)(p);
  u.h[1] = *(const v8h*)(p + 8);
  return u.v;
}

// B fragment (32x16 f16) from pre-swizzled weights: 512 halves/frag, 16 contiguous halves/lane
static __device__ __forceinline__ v16h load_b_frag(const _Float16* w, size_t frag, int lane) {
  const _Float16* p = w + frag * 512 + (size_t)lane * 16;
  union { v16h v; v8h h[2]; } u;
  u.h[0] = *(const v8h*)(p);
  u.h[1] = *(const v8h*)(p + 8);
  return u.v;
}

static __device__ __forceinline__ float sigm(float x) { return 1.f / (1.f + __expf(-x)); }

// ---- weight swizzle: B(k,n) = (k<KA ? A[n,k] : Bsrc[n,k-KA]) -> fragment order fp16 ----
__global__ void swizzle_kernel(_Float16* dst, const float* A, int KA,
                               const float* Bsrc, int KB, int N) {
  int Ktot = KA + KB;
  size_t idx = (size_t)blockIdx.x * blockDim.x + threadIdx.x;
  if (idx >= (size_t)Ktot * N) return;
  int k = (int)(idx % Ktot);
  int n = (int)(idx / Ktot);
  float v = (k < KA) ? A[(size_t)n * KA + k] : Bsrc[(size_t)n * KB + (k - KA)];
  int KT = Ktot >> 5;
  int kt = k >> 5, nt = n >> 4;
  int kl = k & 31, nl = n & 15;
  int lane = nl + ((kl >= 16) ? 16 : 0);
  int r    = (kl & 15) >> 1;
  int half = kl & 1;
  dst[((size_t)(nt * KT + kt)) * 512 + lane * 16 + r * 2 + half] = (_Float16)v;
}

// ---- LayerNorm over [T,DIM] per sample, emit fp16 xn ----
__global__ __launch_bounds__(256) void ln_kernel(const float* x, const float* g,
                                                 const float* bta, _Float16* xn) {
  int b = blockIdx.x, tid = threadIdx.x;
  const float* xb = x + (size_t)b * TT * DIM;
  float s = 0.f, ss = 0.f;
  for (int i = tid; i < TT * DIM; i += 256) { float v = xb[i]; s += v; ss += v * v; }
  __shared__ float r0[256], r1[256];
  r0[tid] = s; r1[tid] = ss; __syncthreads();
  for (int st = 128; st > 0; st >>= 1) {
    if (tid < st) { r0[tid] += r0[tid + st]; r1[tid] += r1[tid + st]; }
    __syncthreads();
  }
  float mu   = r0[0] * (1.f / (TT * DIM));
  float var  = r1[0] * (1.f / (TT * DIM)) - mu * mu;
  float rstd = rsqrtf(var + 1e-5f);
  _Float16* ob = xn + (size_t)b * TT * DIM;
  for (int i = tid; i < TT * DIM; i += 256)
    ob[i] = (_Float16)((xb[i] - mu) * rstd * g[i] + bta[i]);
}

// ---- embed: xe = relu(xn @ We^T + be), WMMA M=16 row tiles ----
__global__ __launch_bounds__(256) void embed_kernel(const _Float16* xn, const _Float16* We_s,
                                                    const float* be, _Float16* xe) {
  int lane = threadIdx.x & 31;
  int wave = threadIdx.x >> 5;
  size_t row0 = (size_t)blockIdx.x * 16;
  const _Float16* abase = xn + row0 * DIM;
  for (int nt = wave; nt < DZ / 16; nt += 8) {
    v8f acc = {};
#pragma unroll
    for (int kt = 0; kt < DIM / 32; ++kt) {
      v16h a = load_a_frag(abase, DIM, kt * 32, lane);
      v16h b = load_b_frag(We_s, (size_t)nt * (DIM / 32) + kt, lane);
      acc = wmma16(a, b, acc);
    }
    int n0 = nt * 16 + (lane & 15);
    int mb = (lane >> 4) * 8;
    float bias = be[n0];
#pragma unroll
    for (int r = 0; r < 8; ++r) {
      float v = acc[r] + bias;
      v = v > 0.f ? v : 0.f;
      xe[(row0 + mb + r) * DZ + n0] = (_Float16)v;
    }
  }
}

// ---- persistent masked LSTM over a batch tile of 16; dir=1 gathers reversed prefix ----
__global__ __launch_bounds__(512) void enc_lstm_kernel(const _Float16* xe, const _Float16* Wcat,
                                                       const float* bih, const float* bhh,
                                                       const int* lengths, float* out_sum, int dir) {
  __shared__ __align__(16) _Float16 stag[16 * 768]; // [m][ h(512) | x(256) ] fp16
  __shared__ float bsum[G4];
  __shared__ int   lens[16];
  int tid = threadIdx.x, lane = tid & 31, wave = tid >> 5;
  int b0 = blockIdx.x * 16;
  for (int i = tid; i < G4; i += 512) bsum[i] = bih[i] + bhh[i];
  for (int i = tid; i < 16 * 768; i += 512) stag[i] = (_Float16)0.f;
  if (tid < 16) lens[tid] = lengths[b0 + tid];
  float c[16], hs[16];
#pragma unroll
  for (int q = 0; q < 16; ++q) { c[q] = 0.f; hs[q] = 0.f; }
  __syncthreads();
  const int KT = 768 / 32;
  for (int t = 0; t < TT; ++t) {
    { // stage x_t (wave w loads row w; 16B per lane)
      int len = lens[wave];
      int ttx = dir ? (len - 1 - t) : t;
      if (ttx < 0) ttx = 0;
      const _Float16* src = xe + ((size_t)(b0 + wave) * TT + ttx) * DZ + lane * 8;
      *(v8h*)&stag[wave * 768 + 512 + lane * 8] = *(const v8h*)src;
    }
    __syncthreads();
    // gates: wave owns tiles {2w,2w+1} (+32,+64,+96) so i/f/g/o stay in-register
    v8f acc[8] = {};
    for (int kt = 0; kt < KT; ++kt) {
      v16h a = load_a_frag(stag, 768, kt * 32, lane);
      if (kt + 1 < KT)
        __builtin_prefetch(Wcat + ((size_t)(2 * wave) * KT + kt + 1) * 512, 0, 1);
#pragma unroll
      for (int i = 0; i < 8; ++i) {
        int nt = 2 * wave + (i & 1) + (i >> 1) * 32;
        v16h b = load_b_frag(Wcat, (size_t)nt * KT + kt, lane);
        acc[i] = wmma16(a, b, acc[i]);
      }
    }
    __syncthreads();
#pragma unroll
    for (int p = 0; p < 2; ++p) {
      int j = 32 * wave + 16 * p + (lane & 15);
      float bi = bsum[j], bf = bsum[512 + j], bg = bsum[1024 + j], bo = bsum[1536 + j];
#pragma unroll
      for (int r = 0; r < 8; ++r) {
        int m = r + 8 * (lane >> 4);
        float c2 = sigm(acc[2 + p][r] + bf) * c[p * 8 + r] +
                   sigm(acc[p][r] + bi) * tanhf(acc[4 + p][r] + bg);
        float h2 = sigm(acc[6 + p][r] + bo) * tanhf(c2);
        if (t < lens[m]) {              // packed-sequence mask: freeze state past len
          c[p * 8 + r] = c2;
          hs[p * 8 + r] += h2;          // masked sum for pooling
          stag[m * 768 + j] = (_Float16)h2;
        }
      }
    }
    __syncthreads();
  }
#pragma unroll
  for (int p = 0; p < 2; ++p) {
    int j = 32 * wave + 16 * p + (lane & 15);
#pragma unroll
    for (int r = 0; r < 8; ++r) {
      int m = r + 8 * (lane >> 4);
      out_sum[(size_t)(b0 + m) * (2 * HH) + dir * HH + j] = hs[p * 8 + r];
    }
  }
}

// ---- pooled latent chain (tiny GEMVs, VALU) ----
__global__ __launch_bounds__(256) void latent_kernel(const float* out_sum, const int* lengths,
    const float* W1e, const float* b1e, const float* W2e, const float* b2e,
    const float* Wd,  const float* bd,  const float* Wp1, const float* bp1,
    const float* Wp2, const float* bp2, float* x0buf, float* pose0buf) {
  int b = blockIdx.x, tid = threadIdx.x;
  __shared__ float vin[1024], v1[512], v2[256], v3[256], v4[256];
  float invlen = 1.f / (float)lengths[b];
  for (int i = tid; i < 1024; i += 256) vin[i] = out_sum[(size_t)b * 1024 + i] * invlen;
  __syncthreads();
  for (int o = tid; o < 512; o += 256) {
    float s = b1e[o]; const float* w = W1e + (size_t)o * 1024;
    for (int k = 0; k < 1024; ++k) s += vin[k] * w[k];
    v1[o] = fmaxf(s, 0.f);
  }
  __syncthreads();
  { int o = tid; float s = b2e[o]; const float* w = W2e + (size_t)o * 512;
    for (int k = 0; k < 512; ++k) s += v1[k] * w[k];
    v2[o] = s; }
  __syncthreads();
  { int o = tid; float s = bd[o]; const float* w = Wd + (size_t)o * 256;
    for (int k = 0; k < 256; ++k) s += v2[k] * w[k];
    float r = fmaxf(s, 0.f);
    v3[o] = r; x0buf[(size_t)b * DZ + o] = r; }
  __syncthreads();
  { int o = tid; float s = bp1[o]; const float* w = Wp1 + (size_t)o * 256;
    for (int k = 0; k < 256; ++k) s += v3[k] * w[k];
    v4[o] = fmaxf(s, 0.f); }
  __syncthreads();
  if (tid < 128) { int o = tid; float s = bp2[o]; const float* w = Wp2 + (size_t)o * 256;
    for (int k = 0; k < 256; ++k) s += v4[k] * w[k];
    pose0buf[(size_t)b * DIM + o] = s; }
}

// ---- decoder: LSTM(K=896) + 2 MLP WMMA stages with pose feedback ----
__global__ __launch_bounds__(512) void dec_lstm_kernel(const _Float16* Wcat, const float* bih,
                                                       const float* bhh,
                                                       const _Float16* W1s, const float* b1d,
                                                       const _Float16* W2s, const float* b2d,
                                                       const float* x0buf, const float* pose0buf,
                                                       float* dec) {
  __shared__ __align__(16) _Float16 stag[16 * 896]; // [m][ h(512) | x0(256) | pose(128) ]
  __shared__ __align__(16) _Float16 hmid[16 * 256];
  __shared__ float bsum[G4];
  int tid = threadIdx.x, lane = tid & 31, wave = tid >> 5;
  int b0 = blockIdx.x * 16;
  for (int i = tid; i < G4; i += 512) bsum[i] = bih[i] + bhh[i];
  for (int i = tid; i < 16 * 896; i += 512) {
    int m = i / 896, k = i % 896;
    float v = 0.f;
    if (k >= 768)      v = pose0buf[(size_t)(b0 + m) * DIM + (k - 768)];
    else if (k >= 512) v = x0buf[(size_t)(b0 + m) * DZ + (k - 512)];
    stag[i] = (_Float16)v;
  }
  float c[16];
#pragma unroll
  for (int q = 0; q < 16; ++q) c[q] = 0.f;
  __syncthreads();
  const int KT = 896 / 32;
  for (int t = 0; t < TT; ++t) {
    v8f acc[8] = {};
    for (int kt = 0; kt < KT; ++kt) {
      v16h a = load_a_frag(stag, 896, kt * 32, lane);
      if (kt + 1 < KT)
        __builtin_prefetch(Wcat + ((size_t)(2 * wave) * KT + kt + 1) * 512, 0, 1);
#pragma unroll
      for (int i = 0; i < 8; ++i) {
        int nt = 2 * wave + (i & 1) + (i >> 1) * 32;
        v16h b = load_b_frag(Wcat, (size_t)nt * KT + kt, lane);
        acc[i] = wmma16(a, b, acc[i]);
      }
    }
    __syncthreads();
#pragma unroll
    for (int p = 0; p < 2; ++p) {
      int j = 32 * wave + 16 * p + (lane & 15);
      float bi = bsum[j], bf = bsum[512 + j], bg = bsum[1024 + j], bo = bsum[1536 + j];
#pragma unroll
      for (int r = 0; r < 8; ++r) {
        int m = r + 8 * (lane >> 4);
        float c2 = sigm(acc[2 + p][r] + bf) * c[p * 8 + r] +
                   sigm(acc[p][r] + bi) * tanhf(acc[4 + p][r] + bg);
        float h2 = sigm(acc[6 + p][r] + bo) * tanhf(c2);
        c[p * 8 + r] = c2;
        stag[m * 896 + j] = (_Float16)h2;
      }
    }
    __syncthreads();
    { // MLP1: relu(h2 @ W1d^T + b1d) -> hmid (K=512, N=256, 1 tile/wave)
      v8f a1 = {};
#pragma unroll 4
      for (int kt = 0; kt < 16; ++kt) {
        v16h a = load_a_frag(stag, 896, kt * 32, lane);
        v16h b = load_b_frag(W1s, (size_t)wave * 16 + kt, lane);
        a1 = wmma16(a, b, a1);
      }
      int n0 = wave * 16 + (lane & 15);
      int mb = (lane >> 4) * 8;
      float bb = b1d[n0];
#pragma unroll
      for (int r = 0; r < 8; ++r) {
        float v = a1[r] + bb;
        v = v > 0.f ? v : 0.f;
        hmid[(mb + r) * 256 + n0] = (_Float16)v;
      }
    }
    __syncthreads();
    if (wave < 8) { // MLP2: pose = hmid @ W2d^T + b2d (K=256, N=128)
      v8f a2 = {};
#pragma unroll
      for (int kt = 0; kt < 8; ++kt) {
        v16h a = load_a_frag(hmid, 256, kt * 32, lane);
        v16h b = load_b_frag(W2s, (size_t)wave * 8 + kt, lane);
        a2 = wmma16(a, b, a2);
      }
      int n0 = wave * 16 + (lane & 15);
      int mb = (lane >> 4) * 8;
      float bb = b2d[n0];
#pragma unroll
      for (int r = 0; r < 8; ++r) {
        float v = a2[r] + bb;
        dec[((size_t)(b0 + mb + r) * TT + t) * DIM + n0] = v;            // output pose
        stag[(mb + r) * 896 + 768 + n0] = (_Float16)v;                   // feedback
      }
    }
    __syncthreads();
  }
}

// ---- pass-through outputs: x and x_lengths ----
__global__ void copy_out_kernel(const float* x, const int* lengths, float* out, size_t nx) {
  size_t i = (size_t)blockIdx.x * 256 + threadIdx.x;
  if (i < nx) out[i] = x[i];
  if (i < BB) ((int*)(out + 2 * nx))[i] = lengths[i];
}

extern "C" void kernel_launch(void* const* d_in, const int* in_sizes, int n_in,
                              void* d_out, int out_size, void* d_ws, size_t ws_size,
                              hipStream_t stream) {
  (void)in_sizes; (void)n_in; (void)out_size; (void)ws_size;
  const float* x      = (const float*)d_in[0];
  const int*   xlen   = (const int*)d_in[1];
  const float* norm_g = (const float*)d_in[2];
  const float* norm_b = (const float*)d_in[3];
  const float* We     = (const float*)d_in[4];
  const float* be     = (const float*)d_in[5];
  const float* Wih_f  = (const float*)d_in[6];
  const float* Whh_f  = (const float*)d_in[7];
  const float* bih_f  = (const float*)d_in[8];
  const float* bhh_f  = (const float*)d_in[9];
  const float* Wih_b  = (const float*)d_in[10];
  const float* Whh_b  = (const float*)d_in[11];
  const float* bih_b  = (const float*)d_in[12];
  const float* bhh_b  = (const float*)d_in[13];
  const float* W1e    = (const float*)d_in[14];
  const float* b1e    = (const float*)d_in[15];
  const float* W2e    = (const float*)d_in[16];
  const float* b2e    = (const float*)d_in[17];
  const float* Wd     = (const float*)d_in[18];
  const float* bd     = (const float*)d_in[19];
  const float* Wp1    = (const float*)d_in[20];
  const float* bp1    = (const float*)d_in[21];
  const float* Wp2    = (const float*)d_in[22];
  const float* bp2    = (const float*)d_in[23];
  const float* Wih_d  = (const float*)d_in[24];
  const float* Whh_d  = (const float*)d_in[25];
  const float* bih_d  = (const float*)d_in[26];
  const float* bhh_d  = (const float*)d_in[27];
  const float* W1d    = (const float*)d_in[28];
  const float* b1d    = (const float*)d_in[29];
  const float* W2d    = (const float*)d_in[30];
  const float* b2d    = (const float*)d_in[31];

  char* wp = (char*)d_ws;
  auto carve = [&](size_t bytes) { char* p = wp; wp += (bytes + 255) & ~(size_t)255; return p; };
  _Float16* xn    = (_Float16*)carve((size_t)BB * TT * DIM * 2);
  _Float16* xe    = (_Float16*)carve((size_t)BB * TT * DZ * 2);
  _Float16* We_s  = (_Float16*)carve((size_t)128 * 256 * 2);
  _Float16* Wcf_s = (_Float16*)carve((size_t)768 * G4 * 2);
  _Float16* Wcb_s = (_Float16*)carve((size_t)768 * G4 * 2);
  _Float16* Wcd_s = (_Float16*)carve((size_t)896 * G4 * 2);
  _Float16* W1s   = (_Float16*)carve((size_t)512 * 256 * 2);
  _Float16* W2s   = (_Float16*)carve((size_t)256 * 128 * 2);
  float* out_sum  = (float*)carve((size_t)BB * 2 * HH * 4);
  float* x0buf    = (float*)carve((size_t)BB * DZ * 4);
  float* pose0buf = (float*)carve((size_t)BB * DIM * 4);

  auto nb = [](size_t n) { return (unsigned)((n + 255) / 256); };

  swizzle_kernel<<<nb((size_t)128 * 256), 256, 0, stream>>>(We_s,  We,    128, nullptr, 0,   256);
  swizzle_kernel<<<nb((size_t)768 * G4),  256, 0, stream>>>(Wcf_s, Whh_f, 512, Wih_f,   256, G4);
  swizzle_kernel<<<nb((size_t)768 * G4),  256, 0, stream>>>(Wcb_s, Whh_b, 512, Wih_b,   256, G4);
  swizzle_kernel<<<nb((size_t)896 * G4),  256, 0, stream>>>(Wcd_s, Whh_d, 512, Wih_d,   384, G4);
  swizzle_kernel<<<nb((size_t)512 * 256), 256, 0, stream>>>(W1s,   W1d,   512, nullptr, 0,   256);
  swizzle_kernel<<<nb((size_t)256 * 128), 256, 0, stream>>>(W2s,   W2d,   256, nullptr, 0,   128);

  ln_kernel<<<BB, 256, 0, stream>>>(x, norm_g, norm_b, xn);
  embed_kernel<<<(BB * TT) / 16, 256, 0, stream>>>(xn, We_s, be, xe);
  enc_lstm_kernel<<<BB / 16, 512, 0, stream>>>(xe, Wcf_s, bih_f, bhh_f, xlen, out_sum, 0);
  enc_lstm_kernel<<<BB / 16, 512, 0, stream>>>(xe, Wcb_s, bih_b, bhh_b, xlen, out_sum, 1);
  latent_kernel<<<BB, 256, 0, stream>>>(out_sum, xlen, W1e, b1e, W2e, b2e,
                                        Wd, bd, Wp1, bp1, Wp2, bp2, x0buf, pose0buf);
  float* out = (float*)d_out;
  size_t nx = (size_t)BB * TT * DIM;
  dec_lstm_kernel<<<BB / 16, 512, 0, stream>>>(Wcd_s, bih_d, bhh_d, W1s, b1d, W2s, b2d,
                                               x0buf, pose0buf, out + nx);
  copy_out_kernel<<<nb(nx), 256, 0, stream>>>(x, xlen, out, nx);
}
